// CausalSelfAttention_31860067402185
// MI455X (gfx1250) — matrix operand order repaired and checked by reading, not compiled, not run on hardware
//
#include <hip/hip_runtime.h>
#include <hip/hip_bf16.h>

// ---------------------------------------------------------------------------
// Types for CDNA5 WMMA (wave32, 16x16x32 bf16 -> f32)
// ---------------------------------------------------------------------------
typedef __attribute__((ext_vector_type(16))) __bf16    bf16x16;
typedef __attribute__((ext_vector_type(8)))  float     v8f;
typedef __attribute__((ext_vector_type(4)))  unsigned int uint4v;

union Frag {
    bf16x16        f;
    uint4v         q[2];
    unsigned int   u[8];
    unsigned short s[16];
};

__device__ __forceinline__ v8f wmma_bf16(bf16x16 a, bf16x16 b, v8f c) {
    // D = A(16x32) * B(32x16) + C, f32 accumulate
    return __builtin_amdgcn_wmma_f32_16x16x32_bf16(
        /*neg_a=*/false, a, /*neg_b=*/false, b,
        /*c_mod=*/(short)0, c, /*reuse_a=*/false, /*reuse_b=*/false);
}

__device__ __forceinline__ unsigned short f2bf(float f) {
    unsigned int u = __float_as_uint(f);
    u += 0x7FFFu + ((u >> 16) & 1u);   // round-to-nearest-even
    return (unsigned short)(u >> 16);
}

// ---------------------------------------------------------------------------
// Problem constants
// ---------------------------------------------------------------------------
namespace cfg {
constexpr int B  = 4;
constexpr int T  = 2048;
constexpr int C  = 1024;
constexpr int H  = 16;
constexpr int DH = 64;
constexpr int M  = B * T;        // 8192 tokens
constexpr int N_QKV = 3 * C;     // 3072
}

// ---------------------------------------------------------------------------
// fp32 -> bf16 row-major convert (A-side operand: x)
// ---------------------------------------------------------------------------
__global__ void k_cvt_bf16(const float* __restrict__ src,
                           unsigned short* __restrict__ dst, int n) {
    int i  = blockIdx.x * blockDim.x + threadIdx.x;
    int st = gridDim.x * blockDim.x;
    for (; i < n; i += st) dst[i] = f2bf(src[i]);
}

// ---------------------------------------------------------------------------
// Weight pre-pack into full WMMA B-fragment layout.
// Fragment (kb = 32-wide K block, nt = 16-wide N tile):
//   lane L, word i (i=0..7) = bf16(w[kb*32 + 8*(L>=16) + 2*(i&3) + 16*(i>>2)][nt*16 + L%16])
//                           | bf16(w[ same +1 ][ same ]) << 16
// Storage: wf[((kb*(N/16) + nt)*32 + lane)*8 + i]
// -> each lane's fragment is 32 contiguous bytes: two global_load_b128.
// ---------------------------------------------------------------------------
__global__ void k_pack_w_frag(const float* __restrict__ w,
                              unsigned int* __restrict__ wf, int K, int N) {
    const int NT = N / 16;
    const int total = (K / 32) * NT * 32 * 8;
    int idx = blockIdx.x * blockDim.x + threadIdx.x;
    int st  = gridDim.x * blockDim.x;
    for (; idx < total; idx += st) {
        int i    = idx & 7;
        int lane = (idx >> 3) & 31;
        int rest = idx >> 8;
        int nt   = rest % NT;
        int kb   = rest / NT;
        int k = kb * 32 + 8 * (lane >> 4) + 2 * (i & 3) + 16 * (i >> 2);
        int n = nt * 16 + (lane & 15);
        unsigned int lo = f2bf(w[(size_t)k * N + n]);
        unsigned int hi = f2bf(w[(size_t)(k + 1) * N + n]);
        wf[idx] = lo | (hi << 16);
    }
}

// ---------------------------------------------------------------------------
// Shared GEMM wave body: 64x64 output tile per wave, K stepped by 32.
// A: row-major bf16 [M,K]. Wf: fragment-swizzled weights (see k_pack_w_frag).
// 16 WMMAs per k-step vs 16 b128 loads.
// ---------------------------------------------------------------------------
__device__ __forceinline__ void gemm_tile_64x64(
    const unsigned short* __restrict__ A, const unsigned int* __restrict__ Wf,
    int K, int N, int m0, int n0, int lane, v8f acc[4][4]) {
    const int hi  = lane >> 4;
    const int l16 = lane & 15;
    const int NT  = N / 16;
    const unsigned short* arow[4];
#pragma unroll
    for (int mi = 0; mi < 4; ++mi)
        arow[mi] = A + (size_t)(m0 + 16 * mi + l16) * K + 8 * hi;

    for (int kk = 0; kk < K; kk += 32) {
        Frag a[4];
#pragma unroll
        for (int mi = 0; mi < 4; ++mi) {
            a[mi].q[0] = *(const uint4v*)(arow[mi] + kk);
            a[mi].q[1] = *(const uint4v*)(arow[mi] + kk + 16);
        }
        const int kb = kk >> 5;
#pragma unroll
        for (int nt = 0; nt < 4; ++nt) {
            Frag b;
            const unsigned int* bp =
                Wf + ((size_t)(kb * NT + (n0 >> 4) + nt) * 32 + lane) * 8;
            b.q[0] = *(const uint4v*)bp;
            b.q[1] = *(const uint4v*)(bp + 4);
#pragma unroll
            for (int mi = 0; mi < 4; ++mi)
                acc[mi][nt] = wmma_bf16(a[mi].f, b.f, acc[mi][nt]);
        }
    }
}

// ---------------------------------------------------------------------------
// QKV GEMM: qkv = x @ w_qkv, fused head-split write-out.
//   Q,K -> bf16 [B,H,T,64] row-major (direct A-frag / K-frag b128 loads)
//   V   -> transposed token-pair-packed uint32 [B,H,64(dh),T/2]
//          word(dh,pr) = bf16(v[2pr][dh]) | bf16(v[2pr+1][dh])<<16
//          -> flash V-fragments become two b128 loads.
// Each 64-wide N strip lies entirely inside one of {Q,K,V} and one head.
// ---------------------------------------------------------------------------
__global__ void __launch_bounds__(128) k_gemm_qkv(
    const unsigned short* __restrict__ xb, const unsigned int* __restrict__ wf,
    unsigned short* __restrict__ Qo, unsigned short* __restrict__ Ko,
    unsigned int* __restrict__ Vt) {
    using namespace cfg;
    const int wav  = blockIdx.x * (blockDim.x >> 5) + (threadIdx.x >> 5);
    const int lane = threadIdx.x & 31;
    const int hi   = lane >> 4, l16 = lane & 15;
    const int mtiles = M / 64;                  // 128
    const int strip  = wav / mtiles;            // 0..47
    const int mt     = wav - strip * mtiles;
    const int m0 = mt * 64, n0 = strip * 64;

    v8f acc[4][4] = {};
    gemm_tile_64x64(xb, wf, C, N_QKV, m0, n0, lane, acc);

    const int sec  = n0 >> 10;                  // 0=Q 1=K 2=V
    const int h    = (n0 & 1023) >> 6;
    const int bidx = m0 / T;
    const int t0   = m0 - bidx * T;

    if (sec < 2) {
        unsigned short* dst =
            (sec == 0 ? Qo : Ko) + ((size_t)(bidx * H + h)) * T * DH;
#pragma unroll
        for (int mi = 0; mi < 4; ++mi)
#pragma unroll
            for (int nt = 0; nt < 4; ++nt)
#pragma unroll
                for (int r = 0; r < 8; ++r) {
                    int t  = t0 + 16 * mi + r + 8 * hi;
                    int dh = nt * 16 + l16;
                    dst[(size_t)t * DH + dh] = f2bf(acc[mi][nt][r]);
                }
    } else {
        unsigned int* dst = Vt + ((size_t)(bidx * H + h)) * DH * (T / 2);
#pragma unroll
        for (int mi = 0; mi < 4; ++mi)
#pragma unroll
            for (int nt = 0; nt < 4; ++nt)
#pragma unroll
                for (int j = 0; j < 4; ++j) {
                    int pr = t0 / 2 + 8 * mi + 4 * hi + j;
                    int dh = nt * 16 + l16;
                    unsigned int lo = f2bf(acc[mi][nt][2 * j]);
                    unsigned int hb = f2bf(acc[mi][nt][2 * j + 1]);
                    dst[(size_t)dh * (T / 2) + pr] = lo | (hb << 16);
                }
    }
}

// ---------------------------------------------------------------------------
// Flash attention: one wave owns 16 query rows of one (b,h).
// S(16x32) via 2x2 WMMAs (K-frags: contiguous b128 pairs), online softmax
// (16-lane butterflies), P re-layout through LDS, O += P@V via 4 WMMAs
// (V-frags: contiguous b128 pairs from transposed pair-packed V).
// ---------------------------------------------------------------------------
__global__ void __launch_bounds__(128) k_flash(
    const unsigned short* __restrict__ Q, const unsigned short* __restrict__ Kt,
    const unsigned int* __restrict__ Vt, unsigned short* __restrict__ Y) {
    using namespace cfg;
    __shared__ __align__(16) unsigned short pls[4][16 * 32];   // P tile / wave

    const int wave = threadIdx.x >> 5;
    const int lane = threadIdx.x & 31;
    const int hi = lane >> 4, l16 = lane & 15;
    const int tile  = blockIdx.x * 4 + wave;        // 8192 q-tiles total
    const int bh    = tile >> 7;                    // T/16 = 128 tiles per bh
    const int q0    = (tile & 127) << 4;
    const int batch = bh >> 4, h = bh & 15;

    const unsigned short* Qb = Q  + (size_t)bh * T * DH;
    const unsigned short* Kb = Kt + (size_t)bh * T * DH;
    const unsigned int*   Vb = Vt + (size_t)bh * DH * (T / 2);

    // Q fragments: 16 rows x 64 dh -> two 16x32 A-fragments
    Frag qf[2];
#pragma unroll
    for (int d = 0; d < 2; ++d) {
        const unsigned short* p = Qb + (size_t)(q0 + l16) * DH + 32 * d + 8 * hi;
        qf[d].q[0] = *(const uint4v*)p;
        qf[d].q[1] = *(const uint4v*)(p + 16);
    }

    v8f o[4] = {};
    float mrow[8], lrow[8];
#pragma unroll
    for (int r = 0; r < 8; ++r) { mrow[r] = -3.0e38f; lrow[r] = 0.0f; }

    const float scale = 0.125f;                     // 1/sqrt(64)
    const int kend = q0 + 16;                       // causal bound (exclusive)

    for (int kb = 0; kb < kend; kb += 32) {
        // ---- S = Q K^T for 32 keys ----
        v8f s[2] = {};
#pragma unroll
        for (int t = 0; t < 2; ++t) {
            const int key = kb + 16 * t + l16;
#pragma unroll
            for (int d = 0; d < 2; ++d) {
                // B-frag 32(dh) x 16(keys): lane's 8 packed words are
                // K[key][32d+8hi .. +7] and K[key][32d+16+8hi .. +7] -> two b128.
                Frag kf;
                const unsigned short* kp =
                    Kb + (size_t)key * DH + 32 * d + 8 * hi;
                kf.q[0] = *(const uint4v*)kp;
                kf.q[1] = *(const uint4v*)(kp + 16);
                s[t] = wmma_bf16(qf[d].f, kf.f, s[t]);
            }
        }
        // ---- scale + causal mask ----
#pragma unroll
        for (int t = 0; t < 2; ++t) {
            const int key = kb + 16 * t + l16;
#pragma unroll
            for (int r = 0; r < 8; ++r) {
                float v = s[t][r] * scale;
                if (key > q0 + r + 8 * hi) v = -3.0e38f;
                s[t][r] = v;
            }
        }
        // ---- online softmax update (row = 16-lane half) ----
#pragma unroll
        for (int r = 0; r < 8; ++r) {
            float mx = fmaxf(s[0][r], s[1][r]);
            mx = fmaxf(mx, __shfl_xor(mx, 1, 16));
            mx = fmaxf(mx, __shfl_xor(mx, 2, 16));
            mx = fmaxf(mx, __shfl_xor(mx, 4, 16));
            mx = fmaxf(mx, __shfl_xor(mx, 8, 16));
            const float mn    = fmaxf(mrow[r], mx);
            const float alpha = __expf(mrow[r] - mn);
            const float p0 = __expf(s[0][r] - mn);
            const float p1 = __expf(s[1][r] - mn);
            s[0][r] = p0; s[1][r] = p1;
            float rs = p0 + p1;
            rs += __shfl_xor(rs, 1, 16);
            rs += __shfl_xor(rs, 2, 16);
            rs += __shfl_xor(rs, 4, 16);
            rs += __shfl_xor(rs, 8, 16);
            mrow[r] = mn;
            lrow[r] = lrow[r] * alpha + rs;
#pragma unroll
            for (int n = 0; n < 4; ++n) o[n][r] *= alpha;
        }
        // ---- P (C-layout) -> LDS -> A-fragment layout ----
#pragma unroll
        for (int t = 0; t < 2; ++t)
#pragma unroll
            for (int r = 0; r < 8; ++r)
                pls[wave][(r + 8 * hi) * 32 + 16 * t + l16] = f2bf(s[t][r]);
        asm volatile("s_wait_dscnt 0x0" ::: "memory");
        Frag pf;
        {
            const unsigned short* pp = &pls[wave][l16 * 32 + 8 * hi];
            pf.q[0] = *(const uint4v*)pp;
            pf.q[1] = *(const uint4v*)(pp + 16);
        }
        // ---- O += P @ V  (transposed pair-packed V: b128 pairs) ----
#pragma unroll
        for (int n = 0; n < 4; ++n) {
            Frag vf;
            const unsigned int* vp =
                Vb + (size_t)(16 * n + l16) * (T / 2) + (kb >> 1) + 4 * hi;
            vf.q[0] = *(const uint4v*)vp;
            vf.q[1] = *(const uint4v*)(vp + 8);
            o[n] = wmma_bf16(pf.f, vf.f, o[n]);
        }
    }
    // ---- normalize + write y [B,T,C] bf16 ----
#pragma unroll
    for (int r = 0; r < 8; ++r) {
        const float inv = 1.0f / lrow[r];
        const int t = q0 + r + 8 * hi;
#pragma unroll
        for (int n = 0; n < 4; ++n) {
            const int dh = n * 16 + l16;
            Y[((size_t)batch * T + t) * C + h * DH + dh] = f2bf(o[n][r] * inv);
        }
    }
}

// ---------------------------------------------------------------------------
// Output projection: out = y @ w_proj, fp32 write-out.
// ---------------------------------------------------------------------------
__global__ void __launch_bounds__(128) k_gemm_proj(
    const unsigned short* __restrict__ yb, const unsigned int* __restrict__ wf,
    float* __restrict__ out) {
    using namespace cfg;
    const int wav  = blockIdx.x * (blockDim.x >> 5) + (threadIdx.x >> 5);
    const int lane = threadIdx.x & 31;
    const int hi = lane >> 4, l16 = lane & 15;
    const int mtiles = M / 64;                  // 128
    const int strip  = wav / mtiles;            // 0..15
    const int mt     = wav - strip * mtiles;
    const int m0 = mt * 64, n0 = strip * 64;

    v8f acc[4][4] = {};
    gemm_tile_64x64(yb, wf, C, C, m0, n0, lane, acc);

#pragma unroll
    for (int mi = 0; mi < 4; ++mi)
#pragma unroll
        for (int nt = 0; nt < 4; ++nt)
#pragma unroll
            for (int r = 0; r < 8; ++r) {
                int m = m0 + 16 * mi + r + 8 * hi;
                out[(size_t)m * C + n0 + nt * 16 + l16] = acc[mi][nt][r];
            }
}

// ---------------------------------------------------------------------------
// Launch
// ---------------------------------------------------------------------------
extern "C" void kernel_launch(void* const* d_in, const int* in_sizes, int n_in,
                              void* d_out, int out_size, void* d_ws,
                              size_t ws_size, hipStream_t stream) {
    using namespace cfg;
    const float* x      = (const float*)d_in[0];   // [B,T,C]
    const float* w_qkv  = (const float*)d_in[1];   // [C,3C]
    const float* w_proj = (const float*)d_in[2];   // [C,C]
    float* out = (float*)d_out;

    char* ws = (char*)d_ws;
    size_t off = 0;
    auto take = [&](size_t bytes) {
        char* p = ws + off;
        off = (off + bytes + 255) & ~(size_t)255;
        return p;
    };
    unsigned short* xb    = (unsigned short*)take((size_t)M * C * 2);        // 16 MiB
    unsigned int*   wqkvf = (unsigned int*)  take((size_t)(C / 2) * N_QKV * 4); // 6 MiB
    unsigned int*   wprjf = (unsigned int*)  take((size_t)(C / 2) * C * 4);  // 2 MiB
    unsigned short* Qb    = (unsigned short*)take((size_t)M * C * 2);        // 16 MiB
    unsigned short* Kb    = (unsigned short*)take((size_t)M * C * 2);        // 16 MiB
    unsigned int*   Vt    = (unsigned int*)  take((size_t)M * C / 2 * 4);    // 16 MiB
    unsigned short* Yb    = (unsigned short*)take((size_t)M * C * 2);        // 16 MiB
    (void)ws_size; (void)n_in; (void)in_sizes; (void)out_size;

    // 1. convert x -> bf16
    k_cvt_bf16<<<4096, 256, 0, stream>>>(x, xb, M * C);
    // 2. pre-swizzle weights into B-fragment layout
    k_pack_w_frag<<<2048, 256, 0, stream>>>(w_qkv, wqkvf, C, N_QKV);
    k_pack_w_frag<<<1024, 256, 0, stream>>>(w_proj, wprjf, C, C);
    // 3. fused QKV GEMM + head split: waves = (M/64)*(3C/64) = 6144
    k_gemm_qkv<<<1536, 128, 0, stream>>>(xb, wqkvf, Qb, Kb, Vt);
    // 4. flash attention: B*H*(T/16) = 8192 q-tiles -> 2048 blocks * 4 waves
    k_flash<<<2048, 128, 0, stream>>>(Qb, Kb, Vt, Yb);
    // 5. output projection: waves = (M/64)*(C/64) = 2048 -> 512 blocks
    k_gemm_proj<<<512, 128, 0, stream>>>(Yb, wprjf, out);
}
